// VectorQuantizer_75754633167468
// MI455X (gfx1250) — compile-verified, matrix operand order checked by reading, not run on hardware
//
#include <hip/hip_runtime.h>
#include <hip/hip_bf16.h>

typedef __attribute__((ext_vector_type(16))) __bf16 v16bf;
typedef __attribute__((ext_vector_type(8)))  float  v8f;

#define NUM_EMB   1024
#define DIM       256
#define NROWS     65536            // 64*32*32
#define ROW_TILES (NROWS / 16)     // 4096
#define COL_TILES (NUM_EMB / 16)   // 64
#define CHUNKS    (DIM / 32)       // 8 K-chunks of 32 per 16x16 output tile
#define LDSPITCH  264              // bf16 elements per staged row (256 + 8 pad
                                   // -> lane stride 132 words mod 64 = 4: the
                                   // 16-lane b128 fragment reads cover all 64
                                   // LDS banks exactly twice = minimum cycles)
#define GATHER_ROWS_PER_BLOCK 8
#define GATHER_BLOCKS (NROWS / GATHER_ROWS_PER_BLOCK)  // 8192

// ---------------- workspace layout (bytes) ----------------
// [0,               524288)   codebook bf16 (1024*256*2)
// [524288,          528384)   enorm f32 (1024*4)
// [528384,          790528)   argmin indices i32 (65536*4)
// [790528,          823296)   loss partials f32 (8192*4)
#define WS_CBB    0
#define WS_ENORM  524288
#define WS_IDX    528384
#define WS_PART   790528

// ---------------------------------------------------------------------------
// Kernel 1: codebook f32 -> bf16, plus exact ||e||^2 per code.
// ---------------------------------------------------------------------------
__global__ void __launch_bounds__(256) vq_prep(const float* __restrict__ cb,
                                               unsigned short* __restrict__ cbb,
                                               float* __restrict__ enorm) {
  const int code = blockIdx.x;
  const int tid  = threadIdx.x;
  const float x = cb[(size_t)code * DIM + tid];
  union { unsigned short u; __bf16 b; } cvt;
  cvt.b = (__bf16)x;
  cbb[(size_t)code * DIM + tid] = cvt.u;

  __shared__ float red[DIM];
  red[tid] = x * x;
  __syncthreads();
  for (int s = 128; s > 0; s >>= 1) {
    if (tid < s) red[tid] += red[tid + s];
    __syncthreads();
  }
  if (tid == 0) enorm[code] = red[0];
}

// ---------------------------------------------------------------------------
// Kernel 2: WMMA distance matmul + running argmin.
// Each wave32 register-blocks TWO 16-row tiles of z (A resident in 128 VGPRs)
// so every B fragment pulled from LDS feeds 2 WMMAs. 8 waves (256 threads)
// per block share the codebook tiles through a double-buffered LDS stage.
// score(row, code) = ||e||^2 - 2 * (z . e)   (||z||^2 constant per row)
// ---------------------------------------------------------------------------
__global__ void __launch_bounds__(256) vq_argmin(
    const float* __restrict__ z,
    const unsigned short* __restrict__ cbb,
    const float* __restrict__ enorm,
    int* __restrict__ idxbuf) {
  __shared__ unsigned short btile[2][16 * LDSPITCH];  // 2 x 8448 B
  __shared__ float senorm[NUM_EMB];                   // 4 KB

  const int tid  = threadIdx.x;
  const int lane = tid & 31;
  const int wave = tid >> 5;
  const int R    = (blockIdx.x * 8 + wave) * 32;   // this wave owns rows [R, R+32)
  const int r    = lane & 15;   // row (A) / col (B) within a 16-tile
  const int half = lane >> 4;   // selects K sub-range per ISA layout

  // staging role: thread -> (row, 16-element segment) of the 16x256 B tile
  const int srow = tid >> 4;   // 0..15
  const int sseg = tid & 15;   // 0..15

  // ---- stage ||e||^2 into LDS (read 64x per wave afterwards) ----
  for (int i = tid; i < NUM_EMB; i += 256) senorm[i] = enorm[i];

  // ---- stage B tile 0 ----
  {
    const uint4* g = (const uint4*)(cbb + (size_t)srow * DIM + sseg * 16);
    uint4 a0 = g[0];
    uint4 a1 = g[1];
    uint4* d = (uint4*)(&btile[0][srow * LDSPITCH + sseg * 16]);
    d[0] = a0;
    d[1] = a1;
  }

  // ---- Load A: two 16x256 f32 tiles -> 16 resident bf16 fragments ----
  v16bf a0f[CHUNKS], a1f[CHUNKS];
  const float* zrow0 = z + (size_t)(R + r) * DIM;
  const float* zrow1 = z + (size_t)(R + 16 + r) * DIM;
#pragma unroll
  for (int c = 0; c < CHUNKS; ++c) {
    const int k0 = c * 32 + half * 8;       // first 8-element run
    const int k1 = k0 + 16;                 // second 8-element run
    float4 f0 = *(const float4*)(zrow0 + k0);
    float4 f1 = *(const float4*)(zrow0 + k0 + 4);
    float4 f2 = *(const float4*)(zrow0 + k1);
    float4 f3 = *(const float4*)(zrow0 + k1 + 4);
    float4 g0 = *(const float4*)(zrow1 + k0);
    float4 g1 = *(const float4*)(zrow1 + k0 + 4);
    float4 g2 = *(const float4*)(zrow1 + k1);
    float4 g3 = *(const float4*)(zrow1 + k1 + 4);
    float t0[16] = {f0.x, f0.y, f0.z, f0.w, f1.x, f1.y, f1.z, f1.w,
                    f2.x, f2.y, f2.z, f2.w, f3.x, f3.y, f3.z, f3.w};
    float t1[16] = {g0.x, g0.y, g0.z, g0.w, g1.x, g1.y, g1.z, g1.w,
                    g2.x, g2.y, g2.z, g2.w, g3.x, g3.y, g3.z, g3.w};
    v16bf av0, av1;
#pragma unroll
    for (int i = 0; i < 16; ++i) { av0[i] = (__bf16)t0[i]; av1[i] = (__bf16)t1[i]; }
    a0f[c] = av0;
    a1f[c] = av1;
  }

  float minv0[8], minv1[8];
  int   mini0[8], mini1[8];
#pragma unroll
  for (int j = 0; j < 8; ++j) {
    minv0[j] = 3.4e38f; mini0[j] = 0;
    minv1[j] = 3.4e38f; mini1[j] = 0;
  }

  __syncthreads();   // tile 0 staged

  for (int ct = 0; ct < COL_TILES; ++ct) {
    const int C = ct * 16;
    const bool hasNext = (ct + 1) < COL_TILES;   // uniform branch

    // prefetch next B tile into registers (loads issue ahead of the WMMAs)
    uint4 p0, p1;
    if (hasNext) {
      const uint4* g =
          (const uint4*)(cbb + (size_t)(C + 16 + srow) * DIM + sseg * 16);
      p0 = g[0];
      p1 = g[1];
    }

    // ---- 16 WMMAs (2 M-tiles) on the current LDS-staged tile ----
    const unsigned short* lrow = &btile[ct & 1][r * LDSPITCH];
    v8f acc0 = {0.f, 0.f, 0.f, 0.f, 0.f, 0.f, 0.f, 0.f};
    v8f acc1 = {0.f, 0.f, 0.f, 0.f, 0.f, 0.f, 0.f, 0.f};
#pragma unroll
    for (int c = 0; c < CHUNKS; ++c) {
      const int k0 = c * 32 + half * 8;
      union { uint4 u[2]; v16bf v; } bb;
      bb.u[0] = *(const uint4*)(lrow + k0);
      bb.u[1] = *(const uint4*)(lrow + k0 + 16);
      // (neg_a, A, neg_b, B, c_mod, C, reuse_a, reuse_b)
      acc0 = __builtin_amdgcn_wmma_f32_16x16x32_bf16(
          false, a0f[c], false, bb.v, (short)0, acc0, false, false);
      acc1 = __builtin_amdgcn_wmma_f32_16x16x32_bf16(
          false, a1f[c], false, bb.v, (short)0, acc1, false, false);
    }

    const float en = senorm[C + r];   // per-lane column norm, shared by all j
#pragma unroll
    for (int j = 0; j < 8; ++j) {
      // D layout: VGPR j, lanes 0-15 -> row M=j, lanes 16-31 -> row M=j+8;
      // column N = lane & 15.
      const float s0 = __builtin_fmaf(-2.0f, acc0[j], en);
      if (s0 < minv0[j]) { minv0[j] = s0; mini0[j] = C + r; }
      const float s1 = __builtin_fmaf(-2.0f, acc1[j], en);
      if (s1 < minv1[j]) { minv1[j] = s1; mini1[j] = C + r; }
    }

    // ---- commit the prefetched tile to the other LDS buffer ----
    if (hasNext) {
      uint4* d = (uint4*)(&btile[(ct + 1) & 1][srow * LDSPITCH + sseg * 16]);
      d[0] = p0;
      d[1] = p1;
    }
    __syncthreads();
  }

  // ---- reduce across the 16 lanes of each half-wave (rows differ by half) ----
#pragma unroll
  for (int m = 1; m < 16; m <<= 1) {
#pragma unroll
    for (int j = 0; j < 8; ++j) {
      const float ov0 = __shfl_xor(minv0[j], m, 32);
      const int   oi0 = __shfl_xor(mini0[j], m, 32);
      if (ov0 < minv0[j] || (ov0 == minv0[j] && oi0 < mini0[j])) {
        minv0[j] = ov0; mini0[j] = oi0;
      }
      const float ov1 = __shfl_xor(minv1[j], m, 32);
      const int   oi1 = __shfl_xor(mini1[j], m, 32);
      if (ov1 < minv1[j] || (ov1 == minv1[j] && oi1 < mini1[j])) {
        minv1[j] = ov1; mini1[j] = oi1;
      }
    }
  }
  if (r == 0) {      // lane 0 holds rows 0..7, lane 16 holds rows 8..15
#pragma unroll
    for (int j = 0; j < 8; ++j) {
      idxbuf[R + half * 8 + j]      = mini0[j];
      idxbuf[R + 16 + half * 8 + j] = mini1[j];
    }
  }
}

// ---------------------------------------------------------------------------
// Kernel 3: gather exact f32 codebook rows -> output, partial MSE sums.
// ---------------------------------------------------------------------------
__global__ void __launch_bounds__(256) vq_gather(const float* __restrict__ z,
                                                 const float* __restrict__ cb,
                                                 const int* __restrict__ idxbuf,
                                                 float* __restrict__ outq,
                                                 float* __restrict__ outIdx,
                                                 float* __restrict__ partials) {
  const int tid = threadIdx.x;
  float acc = 0.f;
#pragma unroll
  for (int rr = 0; rr < GATHER_ROWS_PER_BLOCK; ++rr) {
    const int row = blockIdx.x * GATHER_ROWS_PER_BLOCK + rr;
    const int idx = idxbuf[row];
    const float q  = cb[(size_t)idx * DIM + tid];
    const float zv = z[(size_t)row * DIM + tid];
    outq[(size_t)row * DIM + tid] = q;   // straight-through == quantized
    const float d = q - zv;
    acc = __builtin_fmaf(d, d, acc);
    if (tid == 0) outIdx[row] = (float)idx;
  }
  __shared__ float red[256];
  red[tid] = acc;
  __syncthreads();
  for (int s = 128; s > 0; s >>= 1) {
    if (tid < s) red[tid] += red[tid + s];
    __syncthreads();
  }
  if (tid == 0) partials[blockIdx.x] = red[0];
}

// ---------------------------------------------------------------------------
// Kernel 4: deterministic final reduction -> vq_loss scalar.
// ---------------------------------------------------------------------------
__global__ void __launch_bounds__(256) vq_finish(const float* __restrict__ partials,
                                                 float* __restrict__ outLoss) {
  const int tid = threadIdx.x;
  float acc = 0.f;
  for (int i = tid; i < GATHER_BLOCKS; i += 256) acc += partials[i];
  __shared__ float red[256];
  red[tid] = acc;
  __syncthreads();
  for (int s = 128; s > 0; s >>= 1) {
    if (tid < s) red[tid] += red[tid + s];
    __syncthreads();
  }
  if (tid == 0) {
    // vq_loss = e_latent + 0.25*q_latent; both equal mean((q-z)^2)
    outLoss[0] = red[0] * 1.25f / ((float)NROWS * (float)DIM);
  }
}

// ---------------------------------------------------------------------------
extern "C" void kernel_launch(void* const* d_in, const int* in_sizes, int n_in,
                              void* d_out, int out_size, void* d_ws, size_t ws_size,
                              hipStream_t stream) {
  (void)in_sizes; (void)n_in; (void)out_size; (void)ws_size;
  const float* z  = (const float*)d_in[0];   // 64*32*32*256 f32
  const float* cb = (const float*)d_in[1];   // 1024*256 f32

  char* ws = (char*)d_ws;
  unsigned short* cbb   = (unsigned short*)(ws + WS_CBB);
  float*          enorm = (float*)(ws + WS_ENORM);
  int*            idxb  = (int*)(ws + WS_IDX);
  float*          part  = (float*)(ws + WS_PART);

  float* outq    = (float*)d_out;                                 // [N*DIM]
  float* outLoss = outq + (size_t)NROWS * DIM;                    // [1]
  float* outIdx  = outLoss + 1;                                   // [N]

  vq_prep<<<NUM_EMB, 256, 0, stream>>>(cb, cbb, enorm);
  vq_argmin<<<ROW_TILES / 16, 256, 0, stream>>>(z, cbb, enorm, idxb);
  vq_gather<<<GATHER_BLOCKS, 256, 0, stream>>>(z, cb, idxb, outq, outIdx, part);
  vq_finish<<<1, 256, 0, stream>>>(part, outLoss);
}
